// spline_28836410425489
// MI455X (gfx1250) — compile-verified, matrix operand order checked by reading, not run on hardware
//
#include <hip/hip_runtime.h>
#include <math.h>

#define NN 4096
#define BB 8
#define EE 65536
#define EPSB 1e-5f

typedef __attribute__((ext_vector_type(16))) _Float16 v16h;
typedef __attribute__((ext_vector_type(8)))  _Float16 v8h;
typedef __attribute__((ext_vector_type(8)))  float    v8f;

// ---------- float atomic min/max via int punning (sign-aware, standard trick) ----------
__device__ __forceinline__ void atomicMaxF(float* a, float v) {
  if (v >= 0.f) atomicMax((int*)a, __float_as_int(v));
  else          atomicMin((unsigned int*)a, __float_as_uint(v));
}
__device__ __forceinline__ void atomicMinF(float* a, float v) {
  if (v >= 0.f) atomicMin((int*)a, __float_as_int(v));
  else          atomicMax((unsigned int*)a, __float_as_uint(v));
}

// ---------- fills / conversions ----------
__global__ void k_fill_f32(float* p, long n, float v) {
  long i = (long)blockIdx.x * blockDim.x + threadIdx.x;
  if (i < n) p[i] = v;
}
__global__ void k_fill_i32(int* p, long n, int v) {
  long i = (long)blockIdx.x * blockDim.x + threadIdx.x;
  if (i < n) p[i] = v;
}
__global__ void k_fill_u8(unsigned char* p, long n, unsigned char v) {
  long i = (long)blockIdx.x * blockDim.x + threadIdx.x;
  if (i < n) p[i] = v;
}
__global__ void k_cvt_h(const float* __restrict__ s, _Float16* __restrict__ d, long n) {
  long i = (long)blockIdx.x * blockDim.x + threadIdx.x;
  if (i < n) d[i] = (_Float16)s[i];
}

// ---------- weight transpose+convert: Wt[k][o][i] = f16(W[k][i][o]), k==27 -> root ----------
__global__ void k_wt(const float* __restrict__ W, const float* __restrict__ root,
                     _Float16* __restrict__ Wt, int Fi, int Fo) {
  long t = (long)blockIdx.x * blockDim.x + threadIdx.x;
  long total = (long)28 * Fi * Fo;
  if (t >= total) return;
  int i = (int)(t % Fi);
  long q = t / Fi;
  int o = (int)(q % Fo);
  int k = (int)(q / Fo);
  float v = (k < 27) ? W[((size_t)k * Fi + i) * Fo + o] : root[(size_t)i * Fo + o];
  Wt[t] = (_Float16)v;
}

// ---------- layer 1 xW (Fi == 1, trivial outer product) ----------
__global__ void k_l1_xw(const float* __restrict__ x, const float* __restrict__ W1,
                        const float* __restrict__ root1, _Float16* __restrict__ xWh) {
  long i = (long)blockIdx.x * blockDim.x + threadIdx.x;
  long total = (long)NN * 28 * 64;
  if (i >= total) return;
  int o = (int)(i % 64);
  long t = i / 64;
  int k = (int)(t % 28);
  int node = (int)(t / 28);
  float w = (k < 27) ? W1[k * 64 + o] : root1[o];
  xWh[i] = (_Float16)(x[node] * w);
}

// ---------- WMMA GEMM: xWh[n][k][o] = sum_i Ah[n,i] * Wt[k,o,i]  (fp32 accum) ----------
// One wave per 16x16 C tile. Per CDNA5 ISA 7.12.2, each lane's A (and, with
// transposed weights, B) fragment is two contiguous 16-byte runs -> b128 loads.
__global__ __launch_bounds__(128) void k_wmma_xw(const _Float16* __restrict__ Ah,
                                                 const _Float16* __restrict__ Wt,
                                                 _Float16* __restrict__ xWh,
                                                 int Fi, int Fo) {
  const int lane = threadIdx.x & 31;
  const int wave = threadIdx.x >> 5;
  const int row0 = blockIdx.x << 4;
  const int ct   = blockIdx.y * 4 + wave;     // (k, o-tile) flattened
  const int fo_t = Fo >> 4;
  const int k    = ct / fo_t;
  const int o0   = (ct - k * fo_t) << 4;
  const int r     = lane & 15;
  const int khalf = (lane >> 4) << 3;         // 0 or 8
  const _Float16* Arow = Ah + (size_t)(row0 + r) * Fi + khalf;
  const _Float16* Brow = Wt + ((size_t)k * Fo + o0 + r) * Fi + khalf;
  v8f c = {};
  for (int kk = 0; kk < Fi; kk += 32) {
    v8h alo = *(const v8h*)(Arow + kk);
    v8h ahi = *(const v8h*)(Arow + kk + 16);
    v8h blo = *(const v8h*)(Brow + kk);
    v8h bhi = *(const v8h*)(Brow + kk + 16);
    v16h a = __builtin_shufflevector(alo, ahi, 0, 1, 2, 3, 4, 5, 6, 7,
                                     8, 9, 10, 11, 12, 13, 14, 15);
    v16h b = __builtin_shufflevector(blo, bhi, 0, 1, 2, 3, 4, 5, 6, 7,
                                     8, 9, 10, 11, 12, 13, 14, 15);
    c = __builtin_amdgcn_wmma_f32_16x16x32_f16(false, a, false, b, (short)0, c, false, false);
  }
  const int mb = (lane >> 4) << 3;
  _Float16* out = xWh + ((size_t)(row0 + mb) * 28 + k) * Fo + o0 + r;
  const size_t stride = (size_t)28 * Fo;
#pragma unroll
  for (int v = 0; v < 8; ++v) {
    *out = (_Float16)c[v];
    out += stride;
  }
}

// ---------- edge degree ----------
__global__ void k_edge_cnt(const int* __restrict__ dst, const unsigned char* __restrict__ em,
                           float* __restrict__ cnt) {
  int e = blockIdx.x * blockDim.x + threadIdx.x;
  if (e < EE && em[e]) atomicAdd(&cnt[dst[e]], 1.f);
}

// ---------- spline message passing (8-corner trilinear mix of precomputed xW) ----------
__global__ void k_msg(const float* __restrict__ attr, const unsigned char* __restrict__ em,
                      const int* __restrict__ src, const int* __restrict__ dst,
                      const _Float16* __restrict__ xWh, float* __restrict__ ssum, int Fo) {
  long t = (long)blockIdx.x * blockDim.x + threadIdx.x;
  long total = (long)EE * Fo;
  if (t >= total) return;
  int o = (int)(t % Fo);
  long e = t / Fo;
  if (!em[e]) return;
  float v0 = attr[e * 3 + 0] * 2.f, v1 = attr[e * 3 + 1] * 2.f, v2 = attr[e * 3 + 2] * 2.f;
  int b0 = (int)floorf(v0); b0 = b0 < 0 ? 0 : (b0 > 1 ? 1 : b0);
  int b1 = (int)floorf(v1); b1 = b1 < 0 ? 0 : (b1 > 1 ? 1 : b1);
  int b2 = (int)floorf(v2); b2 = b2 < 0 ? 0 : (b2 > 1 ? 1 : b2);
  float f0 = v0 - (float)b0, f1 = v1 - (float)b1, f2 = v2 - (float)b2;
  const _Float16* base = xWh + (size_t)src[e] * 28 * Fo + o;
  float acc = 0.f;
#pragma unroll
  for (int s = 0; s < 8; ++s) {
    int bit0 = s & 1, bit1 = (s >> 1) & 1, bit2 = (s >> 2) & 1;
    int kidx = ((b0 + bit0) * 3 + (b1 + bit1)) * 3 + (b2 + bit2);
    float w = (bit0 ? f0 : 1.f - f0) * (bit1 ? f1 : 1.f - f1) * (bit2 ? f2 : 1.f - f2);
    acc += w * (float)base[(size_t)kidx * Fo];
  }
  atomicAdd(&ssum[(size_t)dst[e] * Fo + o], acc);
}

// ---------- node update: mean aggr + root + bias -> ELU -> BN(eval) ----------
__global__ void k_node(const float* __restrict__ ssum, const float* __restrict__ cnt,
                       const _Float16* __restrict__ xWh, const float* __restrict__ bias,
                       const float* __restrict__ gam, const float* __restrict__ bet,
                       const float* __restrict__ rm, const float* __restrict__ rv,
                       float* __restrict__ h, int Fo) {
  long t = (long)blockIdx.x * blockDim.x + threadIdx.x;
  long total = (long)NN * Fo;
  if (t >= total) return;
  int o = (int)(t % Fo);
  int n = (int)(t / Fo);
  float pre = ssum[t] / fmaxf(cnt[n], 1.f) + (float)xWh[((size_t)n * 28 + 27) * Fo + o] + bias[o];
  float el = pre > 0.f ? pre : expf(pre) - 1.f;
  h[t] = (el - rm[o]) * gam[o] * rsqrtf(rv[o] + EPSB) + bet[o];
}

// ---------- voxel pooling pipeline ----------
__global__ void k_minmax(const float* __restrict__ pos, const int* __restrict__ vmask,
                         float* pmin, float* pmax) {
  int n = blockIdx.x * blockDim.x + threadIdx.x;
  if (n >= NN || !vmask[n]) return;
  for (int d = 0; d < 3; ++d) {
    float v = pos[n * 3 + d];
    atomicMinF(&pmin[d], v);
    atomicMaxF(&pmax[d], v);
  }
}
__global__ void k_vid(const float* __restrict__ pos, const int* __restrict__ batch,
                      const int* __restrict__ vmask, const float* __restrict__ pmin,
                      const float* __restrict__ pmax, float size, int* __restrict__ vid) {
  int n = blockIdx.x * blockDim.x + threadIdx.x;
  if (n >= NN) return;
  if (!vmask[n]) { vid[n] = 10000000 + n; return; }
  int c0 = (int)floorf((pos[n * 3 + 0] - pmin[0]) / size);
  int c1 = (int)floorf((pos[n * 3 + 1] - pmin[1]) / size);
  int c2 = (int)floorf((pos[n * 3 + 2] - pmin[2]) / size);
  int nv0 = (int)floorf((pmax[0] - pmin[0]) / size) + 1;
  int nv1 = (int)floorf((pmax[1] - pmin[1]) / size) + 1;
  int nv2 = (int)floorf((pmax[2] - pmin[2]) / size) + 1;
  (void)nv0;
  vid[n] = ((batch[n] * nv0 + c0) * nv1 + c1) * nv2 + c2;
}
__global__ void k_first(const int* __restrict__ vid, int* __restrict__ first, int* countp) {
  int i = blockIdx.x * blockDim.x + threadIdx.x;
  if (i >= NN) return;
  int v = vid[i];
  int f = 1;
  for (int j = 0; j < i; ++j) if (vid[j] == v) { f = 0; break; }
  first[i] = f;
  if (f && v < 10000000) atomicAdd(countp, 1);
}
__global__ void k_lab(const int* __restrict__ vid, const int* __restrict__ first,
                      int* __restrict__ lab) {
  int i = blockIdx.x * blockDim.x + threadIdx.x;
  if (i >= NN) return;
  int v = vid[i];
  int c = 0;
  for (int j = 0; j < NN; ++j) c += (first[j] && vid[j] < v) ? 1 : 0;
  lab[i] = c;
}
__global__ void k_scat_h(const float* __restrict__ h, const int* __restrict__ lab,
                         float* __restrict__ hnew, int F) {
  long t = (long)blockIdx.x * blockDim.x + threadIdx.x;
  long total = (long)NN * F;
  if (t >= total) return;
  int o = (int)(t % F);
  int n = (int)(t / F);
  atomicMaxF(&hnew[(size_t)lab[n] * F + o], h[t]);
}
__global__ void k_scat_node(const float* __restrict__ pos, const int* __restrict__ batch,
                            const int* __restrict__ lab, float* __restrict__ psum,
                            float* __restrict__ ccnt, int* __restrict__ bnew) {
  int n = blockIdx.x * blockDim.x + threadIdx.x;
  if (n >= NN) return;
  int l = lab[n];
  atomicAdd(&psum[l * 3 + 0], pos[n * 3 + 0]);
  atomicAdd(&psum[l * 3 + 1], pos[n * 3 + 1]);
  atomicAdd(&psum[l * 3 + 2], pos[n * 3 + 2]);
  atomicAdd(&ccnt[l], 1.f);
  atomicMax(&bnew[l], batch[n]);
}
__global__ void k_fin_h(const float* __restrict__ hnew, const int* countp,
                        float* __restrict__ h, int F) {
  long t = (long)blockIdx.x * blockDim.x + threadIdx.x;
  long total = (long)NN * F;
  if (t >= total) return;
  int n = (int)(t / F);
  h[t] = (n < *countp) ? hnew[t] : 0.f;
}
__global__ void k_fin_node(const float* __restrict__ psum, const float* __restrict__ ccnt,
                           const int* __restrict__ bnew, const int* countp,
                           float* __restrict__ pos, int* __restrict__ batch,
                           int* __restrict__ vmask) {
  int n = blockIdx.x * blockDim.x + threadIdx.x;
  if (n >= NN) return;
  int valid = (n < *countp) ? 1 : 0;
  vmask[n] = valid;
  float c = fmaxf(ccnt[n], 1.f);
  pos[n * 3 + 0] = valid ? psum[n * 3 + 0] / c : 0.f;
  pos[n * 3 + 1] = valid ? psum[n * 3 + 1] / c : 0.f;
  pos[n * 3 + 2] = valid ? psum[n * 3 + 2] / c : 0.f;
  batch[n] = valid ? bnew[n] : 0;
}
__global__ void k_remap(int* __restrict__ src, int* __restrict__ dst,
                        unsigned char* __restrict__ em, const int* __restrict__ lab) {
  int e = blockIdx.x * blockDim.x + threadIdx.x;
  if (e >= EE) return;
  int s = lab[src[e]], d = lab[dst[e]];
  src[e] = s; dst[e] = d;
  em[e] = (em[e] && s != d) ? 1 : 0;
}
__global__ void k_emx(const int* __restrict__ src, const int* __restrict__ dst,
                      const unsigned char* __restrict__ em, const float* __restrict__ pos,
                      float* mxp) {
  int e = blockIdx.x * blockDim.x + threadIdx.x;
  if (e >= EE || !em[e]) return;
  int s = src[e], d = dst[e];
  float m = 0.f;
  for (int k = 0; k < 3; ++k) m = fmaxf(m, fabsf(pos[d * 3 + k] - pos[s * 3 + k]));
  atomicMaxF(mxp, m);
}
__global__ void k_eattr(const int* __restrict__ src, const int* __restrict__ dst,
                        const unsigned char* __restrict__ em, const float* __restrict__ pos,
                        const float* mxp, float* __restrict__ attr) {
  int e = blockIdx.x * blockDim.x + threadIdx.x;
  if (e >= EE) return;
  float scale = 2.f * fmaxf(*mxp, 1e-12f);
  int s = src[e], d = dst[e];
  for (int k = 0; k < 3; ++k)
    attr[e * 3 + k] = em[e] ? (pos[d * 3 + k] - pos[s * 3 + k]) / scale + 0.5f : 0.5f;
}

// ---------- final max_pool_x + MLP head ----------
__global__ void k_gstart(const int* __restrict__ lab, const int* __restrict__ batch,
                         const int* __restrict__ vmask, int* __restrict__ gstart) {
  int n = blockIdx.x * blockDim.x + threadIdx.x;
  if (n >= NN || !vmask[n]) return;
  atomicMin(&gstart[batch[n]], lab[n]);
}
__global__ void k_slot(const float* __restrict__ h, const int* __restrict__ lab,
                       const int* __restrict__ batch, const int* __restrict__ vmask,
                       const int* __restrict__ gstart, float* __restrict__ xp) {
  long t = (long)blockIdx.x * blockDim.x + threadIdx.x;
  long total = (long)NN * 512;
  if (t >= total) return;
  int o = (int)(t % 512);
  int n = (int)(t / 512);
  if (!vmask[n]) return;
  int g = batch[n];
  int local = lab[n] - gstart[g];
  if (local < 0 || local >= 64) return;
  atomicMaxF(&xp[((size_t)g * 64 + local) * 512 + o], h[t]);
}
__global__ void k_xpfix(float* __restrict__ xp) {
  long t = (long)blockIdx.x * blockDim.x + threadIdx.x;
  long total = (long)BB * 64 * 512;
  if (t >= total) return;
  if (xp[t] < -1e29f) xp[t] = 0.f;
}
__global__ void k_fc1(const float* __restrict__ xp, const float* __restrict__ w,
                      const float* __restrict__ b, float* __restrict__ hid) {
  int j = blockIdx.x * blockDim.x + threadIdx.x;  // 1024 output cols
  if (j >= 1024) return;
  float acc[BB];
#pragma unroll
  for (int g = 0; g < BB; ++g) acc[g] = b[j];
  for (int i = 0; i < 64 * 512; ++i) {
    float wv = w[(size_t)i * 1024 + j];
#pragma unroll
    for (int g = 0; g < BB; ++g) acc[g] += xp[(size_t)g * 64 * 512 + i] * wv;
  }
#pragma unroll
  for (int g = 0; g < BB; ++g) {
    float a = acc[g];
    hid[g * 1024 + j] = a > 0.f ? a : expf(a) - 1.f;
  }
}
__global__ void k_fc2(const float* __restrict__ hid, const float* __restrict__ w,
                      const float* __restrict__ b, float* __restrict__ out) {
  int t = threadIdx.x;
  if (t >= BB * 2) return;
  int c = t & 1, g = t >> 1;
  float acc = b[c];
  for (int j = 0; j < 1024; ++j) acc += hid[g * 1024 + j] * w[j * 2 + c];
  out[g * 2 + c] = acc;
}

// ---------------------------------------------------------------------------
extern "C" void kernel_launch(void* const* d_in, const int* in_sizes, int n_in,
                              void* d_out, int out_size, void* d_ws, size_t ws_size,
                              hipStream_t stream) {
  (void)in_sizes; (void)n_in; (void)out_size; (void)ws_size;
  const float* x       = (const float*)d_in[0];
  const float* pos_in  = (const float*)d_in[1];
  const int*   batch_in= (const int*)d_in[2];
  const int*   ei      = (const int*)d_in[3];
  const float* attr_in = (const float*)d_in[4];
  const float *W[5], *root[5], *bias[5], *gam[5], *bet[5], *rm[5], *rv[5];
  for (int l = 1; l <= 4; ++l) {
    int b0 = 5 + (l - 1) * 7;
    W[l]    = (const float*)d_in[b0 + 0];
    root[l] = (const float*)d_in[b0 + 1];
    bias[l] = (const float*)d_in[b0 + 2];
    gam[l]  = (const float*)d_in[b0 + 3];
    bet[l]  = (const float*)d_in[b0 + 4];
    rm[l]   = (const float*)d_in[b0 + 5];
    rv[l]   = (const float*)d_in[b0 + 6];
  }
  const float* fcw1 = (const float*)d_in[33];
  const float* fcb1 = (const float*)d_in[34];
  const float* fcw2 = (const float*)d_in[35];
  const float* fcb2 = (const float*)d_in[36];

  // workspace bump allocator
  char* p = (char*)d_ws;
  auto alloc = [&](size_t bytes) -> void* {
    void* r = (void*)p;
    p += (bytes + 255) & ~(size_t)255;
    return r;
  };
  _Float16* xWh = (_Float16*)alloc((size_t)NN * 28 * 512 * 2);
  _Float16* Wh  = (_Float16*)alloc((size_t)28 * 256 * 512 * 2);
  _Float16* xh  = (_Float16*)alloc((size_t)NN * 256 * 2);
  float* h    = (float*)alloc((size_t)NN * 512 * 4);
  float* hnew = (float*)alloc((size_t)NN * 512 * 4);
  float* ssum = (float*)alloc((size_t)NN * 512 * 4);
  float* cnt  = (float*)alloc((size_t)NN * 4);
  float* posc = (float*)alloc((size_t)NN * 3 * 4);
  float* psum = (float*)alloc((size_t)NN * 3 * 4);
  float* ccnt = (float*)alloc((size_t)NN * 4);
  int* batc   = (int*)alloc((size_t)NN * 4);
  int* bnew   = (int*)alloc((size_t)NN * 4);
  int* srcc   = (int*)alloc((size_t)EE * 4);
  int* dstc   = (int*)alloc((size_t)EE * 4);
  unsigned char* em = (unsigned char*)alloc((size_t)EE);
  float* attr = (float*)alloc((size_t)EE * 3 * 4);
  int* vid    = (int*)alloc((size_t)NN * 4);
  int* first  = (int*)alloc((size_t)NN * 4);
  int* lab    = (int*)alloc((size_t)NN * 4);
  int* vmask  = (int*)alloc((size_t)NN * 4);
  float* pmin = (float*)alloc(3 * 4);
  float* pmax = (float*)alloc(3 * 4);
  float* mx   = (float*)alloc(4);
  int* countp = (int*)alloc(4);
  int* gstart = (int*)alloc(BB * 4);
  float* xp   = (float*)alloc((size_t)BB * 64 * 512 * 4);
  float* hid  = (float*)alloc((size_t)BB * 1024 * 4);

  auto cdiv = [](long a, long b) { return (unsigned)((a + b - 1) / b); };

  // ---- init mutable graph state in workspace ----
  hipMemcpyAsync(posc, pos_in, (size_t)NN * 3 * 4, hipMemcpyDeviceToDevice, stream);
  hipMemcpyAsync(batc, batch_in, (size_t)NN * 4, hipMemcpyDeviceToDevice, stream);
  hipMemcpyAsync(srcc, ei, (size_t)EE * 4, hipMemcpyDeviceToDevice, stream);
  hipMemcpyAsync(dstc, ei + EE, (size_t)EE * 4, hipMemcpyDeviceToDevice, stream);
  hipMemcpyAsync(attr, attr_in, (size_t)EE * 3 * 4, hipMemcpyDeviceToDevice, stream);
  k_fill_u8<<<cdiv(EE, 256), 256, 0, stream>>>(em, EE, 1);
  k_fill_i32<<<cdiv(NN, 256), 256, 0, stream>>>(vmask, NN, 1);

  auto conv = [&](int l, int Fi, int Fo) {
    if (l == 1) {
      long tot = (long)NN * 28 * 64;
      k_l1_xw<<<cdiv(tot, 256), 256, 0, stream>>>(x, W[1], root[1], xWh);
    } else {
      long nWt = (long)28 * Fi * Fo;
      k_wt<<<cdiv(nWt, 256), 256, 0, stream>>>(W[l], root[l], Wh, Fi, Fo);
      k_cvt_h<<<cdiv((long)NN * Fi, 256), 256, 0, stream>>>(h, xh, (long)NN * Fi);
      dim3 g(NN / 16, (28 * (Fo / 16)) / 4);
      k_wmma_xw<<<g, 128, 0, stream>>>(xh, Wh, xWh, Fi, Fo);
    }
    k_fill_f32<<<cdiv((long)NN * Fo, 256), 256, 0, stream>>>(ssum, (long)NN * Fo, 0.f);
    k_fill_f32<<<cdiv(NN, 256), 256, 0, stream>>>(cnt, NN, 0.f);
    k_edge_cnt<<<cdiv(EE, 256), 256, 0, stream>>>(dstc, em, cnt);
    long tot = (long)EE * Fo;
    k_msg<<<cdiv(tot, 256), 256, 0, stream>>>(attr, em, srcc, dstc, xWh, ssum, Fo);
    k_node<<<cdiv((long)NN * Fo, 256), 256, 0, stream>>>(ssum, cnt, xWh, bias[l], gam[l],
                                                         bet[l], rm[l], rv[l], h, Fo);
  };

  auto pool_labels = [&](float size) {
    k_fill_f32<<<1, 32, 0, stream>>>(pmin, 3, 1e30f);
    k_fill_f32<<<1, 32, 0, stream>>>(pmax, 3, -1e30f);
    k_minmax<<<cdiv(NN, 256), 256, 0, stream>>>(posc, vmask, pmin, pmax);
    k_vid<<<cdiv(NN, 256), 256, 0, stream>>>(posc, batc, vmask, pmin, pmax, size, vid);
    k_fill_i32<<<1, 32, 0, stream>>>(countp, 1, 0);
    k_first<<<cdiv(NN, 256), 256, 0, stream>>>(vid, first, countp);
    k_lab<<<cdiv(NN, 256), 256, 0, stream>>>(vid, first, lab);
  };

  auto maxpool = [&](int F) {
    k_fill_f32<<<cdiv((long)NN * F, 256), 256, 0, stream>>>(hnew, (long)NN * F, -3.0e38f);
    k_fill_f32<<<cdiv((long)NN * 3, 256), 256, 0, stream>>>(psum, (long)NN * 3, 0.f);
    k_fill_f32<<<cdiv(NN, 256), 256, 0, stream>>>(ccnt, NN, 0.f);
    k_fill_i32<<<cdiv(NN, 256), 256, 0, stream>>>(bnew, NN, -2147483647 - 1);
    k_scat_h<<<cdiv((long)NN * F, 256), 256, 0, stream>>>(h, lab, hnew, F);
    k_scat_node<<<cdiv(NN, 256), 256, 0, stream>>>(posc, batc, lab, psum, ccnt, bnew);
    k_fin_h<<<cdiv((long)NN * F, 256), 256, 0, stream>>>(hnew, countp, h, F);
    k_fin_node<<<cdiv(NN, 256), 256, 0, stream>>>(psum, ccnt, bnew, countp, posc, batc, vmask);
    k_remap<<<cdiv(EE, 256), 256, 0, stream>>>(srcc, dstc, em, lab);
    k_fill_f32<<<1, 32, 0, stream>>>(mx, 1, 0.f);
    k_emx<<<cdiv(EE, 256), 256, 0, stream>>>(srcc, dstc, em, posc, mx);
    k_eattr<<<cdiv(EE, 256), 256, 0, stream>>>(srcc, dstc, em, posc, mx, attr);
  };

  // ---- network ----
  conv(1, 1, 64);
  pool_labels(0.05f); maxpool(64);
  conv(2, 64, 128);
  pool_labels(0.10f); maxpool(128);
  conv(3, 128, 256);
  pool_labels(0.15f); maxpool(256);
  conv(4, 256, 512);
  pool_labels(0.25f);

  // ---- max_pool_x (64 slots per graph) ----
  k_fill_i32<<<1, 32, 0, stream>>>(gstart, BB, 1000000000);
  k_gstart<<<cdiv(NN, 256), 256, 0, stream>>>(lab, batc, vmask, gstart);
  k_fill_f32<<<cdiv((long)BB * 64 * 512, 256), 256, 0, stream>>>(xp, (long)BB * 64 * 512, -3.0e38f);
  k_slot<<<cdiv((long)NN * 512, 256), 256, 0, stream>>>(h, lab, batc, vmask, gstart, xp);
  k_xpfix<<<cdiv((long)BB * 64 * 512, 256), 256, 0, stream>>>(xp);

  // ---- MLP head ----
  k_fc1<<<4, 256, 0, stream>>>(xp, fcw1, fcb1, hid);
  k_fc2<<<1, 32, 0, stream>>>(hid, fcw2, fcb2, (float*)d_out);
}
